// LTX2Attention_52106543235340
// MI455X (gfx1250) — compile-verified, matrix operand order checked by reading, not run on hardware
//
#include <hip/hip_runtime.h>
#include <hip/hip_bf16.h>
#include <cstdint>

// ---------------------------------------------------------------------------
// LTX2Attention for MI455X (gfx1250): bf16 WMMA pipeline
//   B=2, T=2048, D=2048, HEADS=16, DIM_HEAD=128
// Compute-bound (~240 GFLOP vs ~200MB traffic @ 23.3TB/s) => all matmuls via
// v_wmma_f32_16x16x32_bf16. Attention is flash-style (online softmax).
// ---------------------------------------------------------------------------

#define BATCH   2
#define T_SEQ   2048
#define DMODEL  2048
#define NHEADS  16
#define HDIM    128
#define NTOK    (BATCH * T_SEQ)      // 4096 rows

typedef __attribute__((ext_vector_type(16))) __bf16 v16bf;
typedef __attribute__((ext_vector_type(8)))  float  v8f;
typedef int v4i_vs __attribute__((vector_size(16)));

union frag16 {
    v16bf bf;
    uint4 u4[2];
    unsigned short us[16];
};

__device__ __forceinline__ unsigned short f2bf(float f) {
    unsigned int u = __float_as_uint(f);
    unsigned int r = u + 0x7fffu + ((u >> 16) & 1u);   // round-to-nearest-even
    return (unsigned short)(r >> 16);
}

// ---- async global->LDS copy (CDNA5), guarded ------------------------------
#if defined(__has_builtin)
#if __has_builtin(__builtin_amdgcn_global_load_async_to_lds_b128)
#define HAVE_ASYNC_LDS 1
#endif
#endif

__device__ __forceinline__ void copy16_to_lds(void* lds, const void* g) {
#ifdef HAVE_ASYNC_LDS
    __builtin_amdgcn_global_load_async_to_lds_b128(
        (__attribute__((address_space(1))) v4i_vs*)(g),
        (__attribute__((address_space(3))) v4i_vs*)(lds),
        0, 0);
#else
    *(uint4*)lds = *(const uint4*)g;
#endif
}

__device__ __forceinline__ void wait_async_copies() {
#if defined(__has_builtin)
#if __has_builtin(__builtin_amdgcn_s_wait_asynccnt)
    __builtin_amdgcn_s_wait_asynccnt(0);
#endif
#endif
}

// ---------------------------------------------------------------------------
// f32 -> bf16 conversion
// ---------------------------------------------------------------------------
__global__ void cvt_f32_bf16(const float* __restrict__ src,
                             unsigned short* __restrict__ dst, size_t n) {
    size_t i = (size_t)blockIdx.x * blockDim.x + threadIdx.x;
    size_t stride = (size_t)gridDim.x * blockDim.x;
    for (; i < n; i += stride) dst[i] = f2bf(src[i]);
}

// ---------------------------------------------------------------------------
// Tiled bf16 WMMA GEMM:  C[M,N] = A[M,K] @ B[K,N] + bias[N]
//   block = 256 threads (8 waves), tile 128x128, K-step 32
//   EPI 0: write f32 C
//   EPI 1: write f32 C and bf16 C
//   EPI 2: g = acc+bias;  Cfinal = aux * sigmoid(g)   (gated output)
// ---------------------------------------------------------------------------
template <int EPI>
__global__ __launch_bounds__(256) void gemm_bf16(
    const unsigned short* __restrict__ A, const unsigned short* __restrict__ Bm,
    const float* __restrict__ bias, float* __restrict__ Cf,
    unsigned short* __restrict__ Cb, const float* __restrict__ aux,
    float* __restrict__ Cfinal, int M, int N, int K) {
    __shared__ unsigned short As[128][32];   // [m][k] bf16
    __shared__ unsigned short Bt[128][32];   // [n][k] bf16 (transposed)

    const int tid  = threadIdx.x;
    const int wave = tid >> 5;
    const int lane = tid & 31;
    const int h2   = (lane >> 4) & 1;   // half-wave selector (K-block)
    const int ln   = lane & 15;
    const int m0   = blockIdx.y * 128;
    const int n0   = blockIdx.x * 128;

    v8f acc[8];
#pragma unroll
    for (int nt = 0; nt < 8; ++nt)
#pragma unroll
        for (int i = 0; i < 8; ++i) acc[nt][i] = 0.f;

    for (int k0 = 0; k0 < K; k0 += 32) {
        __syncthreads();
        // A tile: 128x32 halves, async b128 copies (512 chunks / 256 threads)
#pragma unroll
        for (int r = 0; r < 2; ++r) {
            int idx = tid * 2 + r;
            int row = idx >> 2;
            int c8  = idx & 3;
            copy16_to_lds(&As[row][c8 * 8],
                          A + (size_t)(m0 + row) * K + k0 + c8 * 8);
        }
        // B tile transposed: read u64 (4 bf16 along n), scatter into Bt[n][k]
#pragma unroll
        for (int r = 0; r < 4; ++r) {
            int idx = tid * 4 + r;       // 0..1023
            int kr  = idx >> 5;          // 0..31
            int nq  = idx & 31;          // 0..31 (x4 cols)
            const unsigned short* g = Bm + (size_t)(k0 + kr) * N + n0 + nq * 4;
            unsigned long long w = *(const unsigned long long*)g;
            Bt[nq * 4 + 0][kr] = (unsigned short)(w);
            Bt[nq * 4 + 1][kr] = (unsigned short)(w >> 16);
            Bt[nq * 4 + 2][kr] = (unsigned short)(w >> 32);
            Bt[nq * 4 + 3][kr] = (unsigned short)(w >> 48);
        }
        if (k0 + 32 < K)  // warm L2 for the next K chunk
            __builtin_prefetch(A + (size_t)(m0 + (tid >> 1)) * K + k0 + 32, 0, 1);
        wait_async_copies();
        __syncthreads();

        // A fragment: lane ln = row within wave tile; K halves per h2
        frag16 a;
        const unsigned short* arow = &As[wave * 16 + ln][0];
        a.u4[0] = *(const uint4*)(arow + h2 * 8);
        a.u4[1] = *(const uint4*)(arow + 16 + h2 * 8);
#pragma unroll
        for (int nt = 0; nt < 8; ++nt) {
            frag16 bfr;
            const unsigned short* brow = &Bt[nt * 16 + ln][0];
            bfr.u4[0] = *(const uint4*)(brow + h2 * 8);
            bfr.u4[1] = *(const uint4*)(brow + 16 + h2 * 8);
            acc[nt] = __builtin_amdgcn_wmma_f32_16x16x32_bf16(
                false, a.bf, false, bfr.bf, (short)0, acc[nt], false, false);
        }
    }

    // Epilogue (C layout: VGPR i -> row i + 8*h2, col ln)
#pragma unroll
    for (int nt = 0; nt < 8; ++nt) {
#pragma unroll
        for (int i = 0; i < 8; ++i) {
            int row = m0 + wave * 16 + i + h2 * 8;
            int col = n0 + nt * 16 + ln;
            float v = acc[nt][i] + bias[col];
            size_t idx = (size_t)row * N + col;
            if (EPI == 0) {
                Cf[idx] = v;
            } else if (EPI == 1) {
                Cf[idx] = v;
                Cb[idx] = f2bf(v);
            } else {
                float s = 1.f / (1.f + __expf(-v));
                Cfinal[idx] = aux[idx] * s;
            }
        }
    }
}

// ---------------------------------------------------------------------------
// RMSNorm (over D) + rotary, f32 in -> bf16 out.  One block per (b,t) row.
// ---------------------------------------------------------------------------
__global__ __launch_bounds__(256) void rmsnorm_rotary(
    const float* __restrict__ X, const float* __restrict__ gvec,
    const float* __restrict__ cosb, const float* __restrict__ sinb,
    unsigned short* __restrict__ out) {
    __shared__ float red[256];
    const int row = blockIdx.x;              // b*T + t
    const int t   = row & (T_SEQ - 1);
    const int tid = threadIdx.x;
    const float* x = X + (size_t)row * DMODEL;

    float v[8];
    float ss = 0.f;
#pragma unroll
    for (int j = 0; j < 8; ++j) {
        v[j] = x[tid * 8 + j];
        ss += v[j] * v[j];
    }
    red[tid] = ss;
    __syncthreads();
    for (int s = 128; s > 0; s >>= 1) {
        if (tid < s) red[tid] += red[tid + s];
        __syncthreads();
    }
    const float rinv = rsqrtf(red[0] * (1.f / (float)DMODEL) + 1e-6f);

#pragma unroll
    for (int j = 0; j < 8; ++j) {
        int d = tid * 8 + j;
        int head = d >> 7, p = d & 127, half = p >> 6, jj = p & 63;
        int partner = (head << 7) + ((1 - half) << 6) + jj;
        float xn = v[j] * rinv * gvec[d];
        float xp = x[partner] * rinv * gvec[partner];
        float c = cosb[t * 64 + jj], s2 = sinb[t * 64 + jj];
        float o = (half == 0) ? (xn * c - s2 * xp) : (xn * c + s2 * xp);
        out[(size_t)row * DMODEL + d] = f2bf(o);
    }
}

// ---------------------------------------------------------------------------
// Flash attention (bf16 WMMA, online softmax).
//   block = 128 threads (4 waves) sharing one (b,h); each wave owns 16 q rows.
//   k/v chunks of 32 staged in LDS (v transposed), scores & p@v via WMMA.
// ---------------------------------------------------------------------------
__global__ __launch_bounds__(128) void flash_attn(
    const unsigned short* __restrict__ qb, const unsigned short* __restrict__ kb,
    const unsigned short* __restrict__ vb, unsigned short* __restrict__ attnb) {
    __shared__ unsigned short kt[32][128];     // [k-row][d]
    __shared__ unsigned short vt[128][32];     // [d][k-row] (transposed)
    __shared__ float pbuf[4][16][32];          // per-wave prob staging

    const int tid  = threadIdx.x;
    const int wave = tid >> 5;
    const int lane = tid & 31;
    const int h2   = (lane >> 4) & 1;
    const int ln   = lane & 15;
    const int h    = blockIdx.y;
    const int b    = blockIdx.z;
    const int qrow0 = blockIdx.x * 64 + wave * 16;

    // Preload the wave's q tile as 4 A-fragments (K = d in steps of 32)
    frag16 aq[4];
    {
        const unsigned short* qrow =
            qb + ((size_t)(b * T_SEQ + qrow0 + ln) * DMODEL) + h * HDIM;
#pragma unroll
        for (int ds = 0; ds < 4; ++ds) {
            const unsigned short* p = qrow + ds * 32;
            aq[ds].u4[0] = *(const uint4*)(p + h2 * 8);
            aq[ds].u4[1] = *(const uint4*)(p + 16 + h2 * 8);
        }
    }

    v8f o[8];
    float m_i[8], l_i[8];
#pragma unroll
    for (int s = 0; s < 8; ++s)
#pragma unroll
        for (int i = 0; i < 8; ++i) o[s][i] = 0.f;
#pragma unroll
    for (int i = 0; i < 8; ++i) { m_i[i] = -1e30f; l_i[i] = 0.f; }

    const float scale = 0.08838834764831845f;  // 1/sqrt(128)

    for (int kc = 0; kc < T_SEQ; kc += 32) {
        __syncthreads();
        // stage kt: 32 rows x 128 d (512 x b128 chunks / 128 threads)
#pragma unroll
        for (int r = 0; r < 4; ++r) {
            int idx  = tid * 4 + r;
            int krow = idx >> 4;
            int c8   = idx & 15;
            copy16_to_lds(&kt[krow][c8 * 8],
                          kb + ((size_t)(b * T_SEQ + kc + krow) * DMODEL) +
                              h * HDIM + c8 * 8);
        }
        // stage vt transposed: read u64 (4 d), scatter to vt[d][k]
#pragma unroll
        for (int r = 0; r < 8; ++r) {
            int idx  = tid * 8 + r;      // 0..1023
            int krow = idx >> 5;
            int dq   = idx & 31;
            const unsigned short* g =
                vb + ((size_t)(b * T_SEQ + kc + krow) * DMODEL) + h * HDIM + dq * 4;
            unsigned long long w = *(const unsigned long long*)g;
            vt[dq * 4 + 0][krow] = (unsigned short)(w);
            vt[dq * 4 + 1][krow] = (unsigned short)(w >> 16);
            vt[dq * 4 + 2][krow] = (unsigned short)(w >> 32);
            vt[dq * 4 + 3][krow] = (unsigned short)(w >> 48);
        }
        wait_async_copies();
        __syncthreads();

        // scores: S[16q x 32k] as two 16x16 C tiles, K = d (4 steps of 32)
        v8f s0, s1;
#pragma unroll
        for (int i = 0; i < 8; ++i) { s0[i] = 0.f; s1[i] = 0.f; }
#pragma unroll
        for (int ds = 0; ds < 4; ++ds) {
            frag16 bk0, bk1;
            const unsigned short* r0 = &kt[ln][ds * 32];
            bk0.u4[0] = *(const uint4*)(r0 + h2 * 8);
            bk0.u4[1] = *(const uint4*)(r0 + 16 + h2 * 8);
            const unsigned short* r1 = &kt[ln + 16][ds * 32];
            bk1.u4[0] = *(const uint4*)(r1 + h2 * 8);
            bk1.u4[1] = *(const uint4*)(r1 + 16 + h2 * 8);
            s0 = __builtin_amdgcn_wmma_f32_16x16x32_bf16(
                false, aq[ds].bf, false, bk0.bf, (short)0, s0, false, false);
            s1 = __builtin_amdgcn_wmma_f32_16x16x32_bf16(
                false, aq[ds].bf, false, bk1.bf, (short)0, s1, false, false);
        }

        // online softmax: row = i + 8*h2, col = ln (+16 for s1)
        float alpha[8];
#pragma unroll
        for (int i = 0; i < 8; ++i) {
            float a0 = s0[i] * scale, a1 = s1[i] * scale;
            float mx = fmaxf(a0, a1);
            for (int msk = 1; msk < 16; msk <<= 1)
                mx = fmaxf(mx, __shfl_xor(mx, msk, 32));
            float mnew = fmaxf(m_i[i], mx);
            float aold = __expf(m_i[i] - mnew);
            float p0 = __expf(a0 - mnew);
            float p1 = __expf(a1 - mnew);
            float rs = p0 + p1;
            for (int msk = 1; msk < 16; msk <<= 1)
                rs += __shfl_xor(rs, msk, 32);
            l_i[i] = l_i[i] * aold + rs;
            m_i[i] = mnew;
            alpha[i] = aold;
            pbuf[wave][i + h2 * 8][ln]      = p0;
            pbuf[wave][i + h2 * 8][ln + 16] = p1;
        }
#pragma unroll
        for (int s = 0; s < 8; ++s)
#pragma unroll
            for (int i = 0; i < 8; ++i) o[s][i] *= alpha[i];

        // re-layout probs C-tile -> A-fragment (16q x 32k), cvt to bf16
        frag16 ap;
        {
            const float* prow = &pbuf[wave][ln][0];
#pragma unroll
            for (int j = 0; j < 8; ++j) {
                ap.us[j]     = f2bf(prow[h2 * 8 + j]);
                ap.us[8 + j] = f2bf(prow[16 + h2 * 8 + j]);
            }
        }

        // o += P @ V : 8 dim-slices of 16
#pragma unroll
        for (int s = 0; s < 8; ++s) {
            frag16 bv;
            const unsigned short* vr = &vt[ln + s * 16][0];
            bv.u4[0] = *(const uint4*)(vr + h2 * 8);
            bv.u4[1] = *(const uint4*)(vr + 16 + h2 * 8);
            o[s] = __builtin_amdgcn_wmma_f32_16x16x32_bf16(
                false, ap.bf, false, bv.bf, (short)0, o[s], false, false);
        }
    }

    // normalize and write bf16 attention output [b, t, h, hd]
    float rinv[8];
#pragma unroll
    for (int i = 0; i < 8; ++i) rinv[i] = 1.f / l_i[i];
#pragma unroll
    for (int s = 0; s < 8; ++s) {
#pragma unroll
        for (int i = 0; i < 8; ++i) {
            int r = i + h2 * 8;
            int dim = ln + s * 16;
            float val = o[s][i] * rinv[i];
            attnb[((size_t)(b * T_SEQ + qrow0 + r) * DMODEL) + h * HDIM + dim] =
                f2bf(val);
        }
    }
}

// ---------------------------------------------------------------------------
// launch
// ---------------------------------------------------------------------------
extern "C" void kernel_launch(void* const* d_in, const int* in_sizes, int n_in,
                              void* d_out, int out_size, void* d_ws,
                              size_t ws_size, hipStream_t stream) {
    const float* x    = (const float*)d_in[0];
    const float* cosb = (const float*)d_in[1];
    const float* sinb = (const float*)d_in[2];
    const float* wq = (const float*)d_in[3];
    const float* bq = (const float*)d_in[4];
    const float* wk = (const float*)d_in[5];
    const float* bk = (const float*)d_in[6];
    const float* wv = (const float*)d_in[7];
    const float* bv = (const float*)d_in[8];
    const float* gq = (const float*)d_in[9];
    const float* gk = (const float*)d_in[10];
    const float* wo = (const float*)d_in[11];
    const float* bo = (const float*)d_in[12];
    const float* wg = (const float*)d_in[13];
    const float* bg = (const float*)d_in[14];

    const size_t nTokD = (size_t)NTOK * DMODEL;   // 8.4M
    const size_t nWD   = (size_t)DMODEL * DMODEL; // 4.2M

    char* p = (char*)d_ws;
    auto carve = [&](size_t bytes) -> void* {
        void* r = (void*)p;
        p += (bytes + 255) & ~(size_t)255;
        return r;
    };
    unsigned short* xb    = (unsigned short*)carve(nTokD * 2);
    unsigned short* wqb   = (unsigned short*)carve(nWD * 2);
    unsigned short* wkb   = (unsigned short*)carve(nWD * 2);
    unsigned short* wvb   = (unsigned short*)carve(nWD * 2);
    unsigned short* wob   = (unsigned short*)carve(nWD * 2);
    unsigned short* wgb   = (unsigned short*)carve(nWD * 2);
    float*          qf    = (float*)carve(nTokD * 4);
    float*          kf    = (float*)carve(nTokD * 4);
    unsigned short* qbf   = (unsigned short*)carve(nTokD * 2);
    unsigned short* kbf   = (unsigned short*)carve(nTokD * 2);
    unsigned short* vbf   = (unsigned short*)carve(nTokD * 2);
    unsigned short* attnb = (unsigned short*)carve(nTokD * 2);
    float*          outf  = (float*)carve(nTokD * 4);  // also v-gemm f32 dump
    unsigned short* outb  = (unsigned short*)carve(nTokD * 2);

    // 1) downconvert x and weights to bf16
    cvt_f32_bf16<<<2048, 256, 0, stream>>>(x, xb, nTokD);
    cvt_f32_bf16<<<1024, 256, 0, stream>>>(wq, wqb, nWD);
    cvt_f32_bf16<<<1024, 256, 0, stream>>>(wk, wkb, nWD);
    cvt_f32_bf16<<<1024, 256, 0, stream>>>(wv, wvb, nWD);
    cvt_f32_bf16<<<1024, 256, 0, stream>>>(wo, wob, nWD);
    cvt_f32_bf16<<<1024, 256, 0, stream>>>(wg, wgb, nWD);

    // 2) q/k/v projections
    dim3 ggrid(DMODEL / 128, NTOK / 128);  // (16, 32)
    gemm_bf16<0><<<ggrid, 256, 0, stream>>>(xb, wqb, bq, qf, nullptr, nullptr,
                                            nullptr, NTOK, DMODEL, DMODEL);
    gemm_bf16<0><<<ggrid, 256, 0, stream>>>(xb, wkb, bk, kf, nullptr, nullptr,
                                            nullptr, NTOK, DMODEL, DMODEL);
    gemm_bf16<1><<<ggrid, 256, 0, stream>>>(xb, wvb, bv, outf, vbf, nullptr,
                                            nullptr, NTOK, DMODEL, DMODEL);

    // 3) rmsnorm + rotary -> bf16 q/k
    rmsnorm_rotary<<<NTOK, 256, 0, stream>>>(qf, gq, cosb, sinb, qbf);
    rmsnorm_rotary<<<NTOK, 256, 0, stream>>>(kf, gk, cosb, sinb, kbf);

    // 4) flash attention
    dim3 fgrid(T_SEQ / 64, NHEADS, BATCH);
    flash_attn<<<fgrid, 128, 0, stream>>>(qbf, kbf, vbf, attnb);

    // 5) output projection (f32 + bf16 copies)
    gemm_bf16<1><<<ggrid, 256, 0, stream>>>(attnb, wob, bo, outf, outb, nullptr,
                                            nullptr, NTOK, DMODEL, DMODEL);

    // 6) gate projection fused with out * sigmoid(gate) -> d_out
    gemm_bf16<2><<<ggrid, 256, 0, stream>>>(outb, wgb, bg, nullptr, nullptr,
                                            outf, (float*)d_out, NTOK, DMODEL,
                                            DMODEL);
}
